// TransformerBlock_13417477833315
// MI455X (gfx1250) — compile-verified, matrix operand order checked by reading, not run on hardware
//
#include <hip/hip_runtime.h>
#include <hip/hip_bf16.h>

// ---------------------------------------------------------------------------
// MI455X / gfx1250, wave32. All heavy math goes through v_wmma_f32_16x16x32_bf16
// (bf16 in, f32 accumulate). Problem constants from the reference.
// ---------------------------------------------------------------------------
#define BB 4
#define CC 256
#define NN 4096          // 16*16*16 tokens
#define HEADS 8
#define DH 32            // head dim
#define PP 64            // proj size

typedef __attribute__((ext_vector_type(16))) __bf16 v16bf;
typedef __attribute__((ext_vector_type(8)))  float  v8f;

__device__ __forceinline__ float bf2f(__bf16 b) {
  unsigned short s = __builtin_bit_cast(unsigned short, b);
  unsigned u = ((unsigned)s) << 16;
  return __builtin_bit_cast(float, u);
}
__device__ __forceinline__ __bf16 f2bf(float f) {
  unsigned u = __builtin_bit_cast(unsigned, f);
  unsigned r = (u + 0x7FFFu + ((u >> 16) & 1u)) >> 16;
  return __builtin_bit_cast(__bf16, (unsigned short)r);
}

// ---------------------------------------------------------------------------
// fp32 -> bf16 conversion (weights)
// ---------------------------------------------------------------------------
__global__ void cvt_f32_to_bf16(const float* __restrict__ in,
                                __bf16* __restrict__ out, long n) {
  long t = (long)blockIdx.x * blockDim.x + threadIdx.x;
  if (t < n) out[t] = f2bf(in[t]);
}

// ---------------------------------------------------------------------------
// LayerNorm over C for each (b,n) token; x is (B,C,N), xn is (B,N,C) bf16
// ---------------------------------------------------------------------------
__global__ void layernorm_kernel(const float* __restrict__ x,
                                 const float* __restrict__ ln_w,
                                 const float* __restrict__ ln_b,
                                 __bf16* __restrict__ xn) {
  long t = (long)blockIdx.x * blockDim.x + threadIdx.x;   // 0..B*N-1
  int b = (int)(t >> 12), n = (int)(t & 4095);
  const float* xp = x + ((long)b * CC) * NN + n;          // stride N per channel
  float s = 0.f, s2 = 0.f;
  for (int c = 0; c < CC; ++c) { float v = xp[(long)c * NN]; s += v; s2 += v * v; }
  float mu = s * (1.f / CC);
  float var = s2 * (1.f / CC) - mu * mu;
  float rstd = rsqrtf(var + 1e-5f);
  __bf16* op = xn + t * CC;
  for (int c = 0; c < CC; ++c) {
    float v = xp[(long)c * NN];
    op[c] = f2bf((v - mu) * rstd * ln_w[c] + ln_b[c]);
  }
}

// ---------------------------------------------------------------------------
// A/B fragment fill helpers (ISA 16-bit fragment layout: lanes = rows/cols,
// VGPR pairs pack K; half-wave covers K offsets {0..7,16..23} / {8..15,24..31}).
// ---------------------------------------------------------------------------
__device__ __forceinline__ v16bf load_frag(const __bf16* __restrict__ row) {
  v16bf f;
#pragma unroll
  for (int v = 0; v < 8; ++v) {
    int kb = ((v & 3) * 2) + ((v >> 2) * 16);
    f[2 * v]     = row[kb];
    f[2 * v + 1] = row[kb + 1];
  }
  return f;
}

// ---------------------------------------------------------------------------
// Generic batched WMMA GEMM: Out[m,n] = sum_k A[m,k]*B[n,k] (+bias)
// A: (M x K) row-major bf16, B: (N x K) row-major bf16 (i.e. computes A*B^T).
// Base kernel: one wave per 16x16 tile. K % 32 == 0, M,N % 16 == 0.
// Output batch offset = (z/oDiv)*sO1 + (z%oDiv)*sO2 (handles (b,h) scatter).
// biasMode: 0 none, 1 bias[n], 2 bias[m].
// ---------------------------------------------------------------------------
__global__ void wmma_gemm_kernel(const __bf16* __restrict__ A,
                                 const __bf16* __restrict__ Bm,
                                 const float* __restrict__ bias,
                                 float* __restrict__ OutF,
                                 __bf16* __restrict__ OutB,
                                 int K, int lda, int ldb, int ldo,
                                 long sA, long sB, long sO1, long sO2, int oDiv,
                                 int biasMode) {
  int lane = threadIdx.x;          // 0..31
  int n0 = blockIdx.x * 16;
  int m0 = blockIdx.y * 16;
  int z = blockIdx.z;
  const __bf16* Ab = A + (long)z * sA;
  const __bf16* Bb = Bm + (long)z * sB;
  long oOff = (long)(z / oDiv) * sO1 + (long)(z % oDiv) * sO2;
  int half = lane >> 4;
  int lr = lane & 15;
  v8f acc = {};
  for (int kk = 0; kk < K; kk += 32) {
    v16bf af = load_frag(Ab + (long)(m0 + lr) * lda + kk + half * 8);
    v16bf bfr = load_frag(Bb + (long)(n0 + lr) * ldb + kk + half * 8);
    acc = __builtin_amdgcn_wmma_f32_16x16x32_bf16(false, af, false, bfr,
                                                  (short)0, acc, false, false);
  }
#pragma unroll
  for (int r = 0; r < 8; ++r) {
    int m = m0 + r + half * 8;
    int n = n0 + lr;
    float v = acc[r];
    if (biasMode == 1) v += bias[n];
    else if (biasMode == 2) v += bias[m];
    long idx = oOff + (long)m * ldo + n;
    if (OutB) OutB[idx] = f2bf(v);
    else      OutF[idx] = v;
  }
}

// ---------------------------------------------------------------------------
// Register-blocked variant: one wave computes a 16x64 tile (4 accumulators),
// A fragment loaded once per K-step and reused for 4 WMMAs (4 wmma : 10 b128).
// Requires N % 64 == 0.
// ---------------------------------------------------------------------------
__global__ void wmma_gemm4_kernel(const __bf16* __restrict__ A,
                                  const __bf16* __restrict__ Bm,
                                  const float* __restrict__ bias,
                                  float* __restrict__ OutF,
                                  __bf16* __restrict__ OutB,
                                  int K, int lda, int ldb, int ldo,
                                  long sA, long sB, long sO1, long sO2, int oDiv,
                                  int biasMode) {
  int lane = threadIdx.x;
  int n0 = blockIdx.x * 64;
  int m0 = blockIdx.y * 16;
  int z = blockIdx.z;
  const __bf16* Ab = A + (long)z * sA;
  const __bf16* Bb = Bm + (long)z * sB;
  long oOff = (long)(z / oDiv) * sO1 + (long)(z % oDiv) * sO2;
  int half = lane >> 4;
  int lr = lane & 15;
  v8f acc[4] = {};
  for (int kk = 0; kk < K; kk += 32) {
    v16bf af = load_frag(Ab + (long)(m0 + lr) * lda + kk + half * 8);
#pragma unroll
    for (int u = 0; u < 4; ++u) {
      v16bf bfr = load_frag(Bb + (long)(n0 + u * 16 + lr) * ldb + kk + half * 8);
      acc[u] = __builtin_amdgcn_wmma_f32_16x16x32_bf16(false, af, false, bfr,
                                                       (short)0, acc[u], false, false);
    }
  }
#pragma unroll
  for (int u = 0; u < 4; ++u) {
#pragma unroll
    for (int r = 0; r < 8; ++r) {
      int m = m0 + r + half * 8;
      int n = n0 + u * 16 + lr;
      float v = acc[u][r];
      if (biasMode == 1) v += bias[n];
      else if (biasMode == 2) v += bias[m];
      long idx = oOff + (long)m * ldo + n;
      if (OutB) OutB[idx] = f2bf(v);
      else      OutF[idx] = v;
    }
  }
}

// ---------------------------------------------------------------------------
// Implicit-im2col WMMA conv: Out(b,co,n) = sum_{ci,kh,kw,kd} In * Wt
// In: (B,Cin,4096) bf16, Wt: (O=256, K=Cin*ks^3) bf16, Out: (B,256,4096) f32.
// Block = 256 threads = 8 waves computing a 64m x 128co macro-tile:
//   - 64x32 im2col slab staged in LDS per K-step (cooperative gather, 8/thread)
//   - wave w: m-tile = w>>1, co-group = (w&1)*64; 4 co-subtiles per wave
// Weight-panel traffic drops 4x vs a 16-row macro-tile; A fragments come from
// LDS (ds_load_b128) and are reused for 4 WMMAs each.
// ---------------------------------------------------------------------------
__global__ void wmma_conv_kernel(const __bf16* __restrict__ In,
                                 const __bf16* __restrict__ Wt,
                                 float* __restrict__ Out,
                                 int Cin, int ks, int pad, int K) {
  __shared__ __bf16 sA[64 * 32];
  int tid = threadIdx.x;
  int wave = tid >> 5, lane = tid & 31;
  int half = lane >> 4, lr = lane & 15;
  int mBlock = blockIdx.y * 64;
  int mt = wave >> 1;                       // wave's m-tile 0..3
  int co0 = blockIdx.x * 128 + (wave & 1) * 64;
  int ks2 = ks * ks, ks3 = ks2 * ks;
  v8f acc[4] = {};
  for (int kk = 0; kk < K; kk += 32) {
    __syncthreads();
    // cooperative im2col gather: 64x32 elements, 8 per thread (one row chunk)
    int e0 = tid * 8;
    int row = e0 >> 5;
    int col0 = e0 & 31;
    int mg = mBlock + row;
    int gb = mg >> 12, gn = mg & 4095;
    int gh = gn >> 8, gw = (gn >> 4) & 15, gd = gn & 15;
#pragma unroll
    for (int j = 0; j < 8; ++j) {
      int kg = kk + col0 + j;
      int ci = kg / ks3; int r = kg - ci * ks3;
      int k0 = r / ks2;  int r2 = r - k0 * ks2;
      int k1 = r2 / ks;  int k2 = r2 - k1 * ks;
      int zh = gh + k0 - pad, zw = gw + k1 - pad, zd = gd + k2 - pad;
      float fv = 0.f;
      if ((unsigned)zh < 16u && (unsigned)zw < 16u && (unsigned)zd < 16u)
        fv = bf2f(In[(((long)gb * Cin + ci) << 12) + zh * 256 + zw * 16 + zd]);
      sA[row * 32 + col0 + j] = f2bf(fv);
    }
    __syncthreads();
    v16bf af = load_frag(sA + (mt * 16 + lr) * 32 + half * 8);
#pragma unroll
    for (int u = 0; u < 4; ++u) {
      v16bf bfr = load_frag(Wt + (long)(co0 + u * 16 + lr) * K + kk + half * 8);
      acc[u] = __builtin_amdgcn_wmma_f32_16x16x32_bf16(false, af, false, bfr,
                                                       (short)0, acc[u], false, false);
    }
  }
#pragma unroll
  for (int u = 0; u < 4; ++u) {
#pragma unroll
    for (int r = 0; r < 8; ++r) {
      int m = mBlock + mt * 16 + r + half * 8;
      int co = co0 + u * 16 + lr;
      int b = m >> 12, n = m & 4095;
      Out[((long)b * 256 + co) * 4096 + n] = acc[u][r];
    }
  }
}

// ---------------------------------------------------------------------------
// 1/||row||2 for q and k rows (over N). qkvv: (B,N,1024) bf16.
// blockIdx.x: [0,1024) -> q rows, [1024,2048) -> k rows, r=(b*8+h)*32+d.
// ---------------------------------------------------------------------------
__global__ void qk_norm_kernel(const __bf16* __restrict__ qkvv,
                               float* __restrict__ rnorm) {
  int row = blockIdx.x;
  int which = row >> 10;
  int r = row & 1023;
  int b = r >> 8, h = (r >> 5) & 7, dd = r & 31;
  const __bf16* p = qkvv + ((long)b * NN) * 1024 + which * 256 + h * 32 + dd;
  __shared__ float red[256];
  float s = 0.f;
  for (int n = threadIdx.x; n < NN; n += 256) {
    float v = bf2f(p[(long)n * 1024]);
    s += v * v;
  }
  red[threadIdx.x] = s; __syncthreads();
  for (int st = 128; st > 0; st >>= 1) {
    if (threadIdx.x < st) red[threadIdx.x] += red[threadIdx.x + st];
    __syncthreads();
  }
  if (threadIdx.x == 0) rnorm[row] = 1.f / fmaxf(sqrtf(red[0]), 1e-12f);
}

// ---------------------------------------------------------------------------
// Repack qkvv (B,N,1024) into per-head WMMA-friendly layouts.
// ---------------------------------------------------------------------------
__global__ void repack_kernel(const __bf16* __restrict__ qkvv,
                              const float* __restrict__ rnorm,
                              __bf16* __restrict__ qnT,   // (B,h,N,d)
                              __bf16* __restrict__ qnN,   // (B,h,d,N)
                              __bf16* __restrict__ knN,   // (B,h,d,N) normalized
                              __bf16* __restrict__ kN,    // (B,h,d,N) raw
                              __bf16* __restrict__ vcaT,  // (B,h,N,d)
                              __bf16* __restrict__ vsaN) {// (B,h,d,N)
  long t = (long)blockIdx.x * blockDim.x + threadIdx.x;   // B*N*C
  int c = (int)(t & 255);
  int n = (int)((t >> 8) & 4095);
  int b = (int)(t >> 20);
  int h = c >> 5, dd = c & 31;
  const __bf16* base = qkvv + ((long)(b * NN + n)) * 1024 + c;
  float q = bf2f(base[0]);
  float k = bf2f(base[256]);
  int r = (b << 8) | (h << 5) | dd;
  float rq = rnorm[r], rk = rnorm[1024 + r];
  long bh = b * HEADS + h;
  long idxT = (bh * NN + n) * DH + dd;
  long idxN = (bh * DH + dd) * NN + n;
  __bf16 qn = f2bf(q * rq);
  qnT[idxT] = qn;
  qnN[idxN] = qn;
  knN[idxN] = f2bf(k * rk);
  kN[idxN]  = base[256];
  vcaT[idxT] = base[512];
  vsaN[idxN] = base[768];
}

// ---------------------------------------------------------------------------
// Channel-attention softmax: logits (B*h,32,32) f32 -> attn_ca bf16, scale temp[h]
// ---------------------------------------------------------------------------
__global__ void softmax_ca_kernel(const float* __restrict__ logits,
                                  const float* __restrict__ temp,
                                  __bf16* __restrict__ out) {
  int row = blockIdx.x * blockDim.x + threadIdx.x;        // 1024 rows of 32
  if (row >= 32 * 32) return;
  int z = row >> 5;
  float tm = temp[z & 7];
  const float* p = logits + (long)row * 32;
  float e[32], mx = -1e30f;
#pragma unroll
  for (int i = 0; i < 32; ++i) { e[i] = p[i] * tm; mx = fmaxf(mx, e[i]); }
  float s = 0.f;
#pragma unroll
  for (int i = 0; i < 32; ++i) { e[i] = __expf(e[i] - mx); s += e[i]; }
  float inv = 1.f / s;
#pragma unroll
  for (int i = 0; i < 32; ++i) out[(long)row * 32 + i] = f2bf(e[i] * inv);
}

// ---------------------------------------------------------------------------
// Spatial-attention softmax in place: attn (B,h,N,64) bf16, scale temp2[h]
// ---------------------------------------------------------------------------
__global__ void softmax_sa_kernel(__bf16* __restrict__ attn,
                                  const float* __restrict__ temp2) {
  long row = (long)blockIdx.x * blockDim.x + threadIdx.x; // B*h*N rows
  int z = (int)(row >> 12);
  float tm = temp2[z & 7];
  __bf16* p = attn + row * PP;
  float e[PP], mx = -1e30f;
#pragma unroll
  for (int i = 0; i < PP; ++i) { e[i] = bf2f(p[i]) * tm; mx = fmaxf(mx, e[i]); }
  float s = 0.f;
#pragma unroll
  for (int i = 0; i < PP; ++i) { e[i] = __expf(e[i] - mx); s += e[i]; }
  float inv = 1.f / s;
#pragma unroll
  for (int i = 0; i < PP; ++i) p[i] = f2bf(e[i] * inv);
}

// ---------------------------------------------------------------------------
// Faithful scrambled reshape of x_sa: (B,h,N,d) -> permute(0,3,1,2) -> (B,N,C)
// ---------------------------------------------------------------------------
__global__ void scramble_kernel(const __bf16* __restrict__ xsa_tmp,
                                __bf16* __restrict__ xsa_s) {
  long t = (long)blockIdx.x * blockDim.x + threadIdx.x;   // B*h*N*d
  int dd = (int)(t & 31);
  int n = (int)((t >> 5) & 4095);
  int h = (int)((t >> 17) & 7);
  int b = (int)(t >> 20);
  long flat = ((long)(dd * HEADS + h)) * NN + n;          // flatten (d,h,n)
  int n2 = (int)(flat >> 8);
  int c2 = (int)(flat & 255);
  xsa_s[((long)b * NN + n2) * CC + c2] = xsa_tmp[t];
}

// ---------------------------------------------------------------------------
// attn_skip = x + gamma * epa, (B,C,N) f32 + bf16 copies
// ---------------------------------------------------------------------------
__global__ void residual_kernel(const float* __restrict__ x,
                                const float* __restrict__ gamma,
                                const float* __restrict__ epa,   // (B,N,C) f32
                                float* __restrict__ skipF,
                                __bf16* __restrict__ skipB) {
  long t = (long)blockIdx.x * blockDim.x + threadIdx.x;   // B*C*N (b,c,n)
  int n = (int)(t & 4095);
  int c = (int)((t >> 12) & 255);
  int b = (int)(t >> 20);
  float v = x[t] + gamma[c] * epa[((long)(b * NN + n)) * CC + c];
  skipF[t] = v;
  skipB[t] = f2bf(v);
}

// ---------------------------------------------------------------------------
// BatchNorm3d training-mode stats (biased var) per channel over (B,N)
// ---------------------------------------------------------------------------
__global__ void bn_stats_kernel(const float* __restrict__ y,
                                float* __restrict__ stats) {
  int c = blockIdx.x;
  __shared__ float rs[256], rs2[256];
  float s = 0.f, s2 = 0.f;
  for (int i = threadIdx.x; i < BB * NN; i += 256) {
    int b = i >> 12, n = i & 4095;
    float v = y[((long)b * CC + c) * NN + n];
    s += v; s2 += v * v;
  }
  rs[threadIdx.x] = s; rs2[threadIdx.x] = s2; __syncthreads();
  for (int st = 128; st > 0; st >>= 1) {
    if (threadIdx.x < st) { rs[threadIdx.x] += rs[threadIdx.x + st];
                            rs2[threadIdx.x] += rs2[threadIdx.x + st]; }
    __syncthreads();
  }
  if (threadIdx.x == 0) {
    float mu = rs[0] / (BB * NN);
    float var = rs2[0] / (BB * NN) - mu * mu;
    stats[2 * c] = mu;
    stats[2 * c + 1] = rsqrtf(var + 1e-5f);
  }
}

// h1 = lrelu(bn1(conv1_out)) -> bf16
__global__ void bn_lrelu_kernel(const float* __restrict__ y,
                                const float* __restrict__ stats,
                                const float* __restrict__ w,
                                const float* __restrict__ bbias,
                                __bf16* __restrict__ out) {
  long t = (long)blockIdx.x * blockDim.x + threadIdx.x;
  int c = (int)((t >> 12) & 255);
  float v = (y[t] - stats[2 * c]) * stats[2 * c + 1] * w[c] + bbias[c];
  v = v >= 0.f ? v : 0.01f * v;
  out[t] = f2bf(v);
}

// convf = lrelu(bn2(conv2_out) + attn_skip) -> bf16
__global__ void bn_add_lrelu_kernel(const float* __restrict__ y,
                                    const float* __restrict__ stats,
                                    const float* __restrict__ w,
                                    const float* __restrict__ bbias,
                                    const float* __restrict__ skipF,
                                    __bf16* __restrict__ out) {
  long t = (long)blockIdx.x * blockDim.x + threadIdx.x;
  int c = (int)((t >> 12) & 255);
  float v = (y[t] - stats[2 * c]) * stats[2 * c + 1] * w[c] + bbias[c] + skipF[t];
  v = v >= 0.f ? v : 0.01f * v;
  out[t] = f2bf(v);
}

// out = attn_skip + conv8_out + conv8_b
__global__ void final_kernel(const float* __restrict__ skipF,
                             const float* __restrict__ c8,
                             const float* __restrict__ b8,
                             float* __restrict__ out) {
  long t = (long)blockIdx.x * blockDim.x + threadIdx.x;
  int c = (int)((t >> 12) & 255);
  out[t] = skipF[t] + c8[t] + b8[c];
}

// ---------------------------------------------------------------------------
// Workspace layout (phase-aliased, ~145 MB total)
// ---------------------------------------------------------------------------
static constexpr size_t SZ_SKIPF   = (size_t)BB * CC * NN * 4;      // 16 MB f32
static constexpr size_t SZ_SKIPB   = SZ_SKIPF / 2;                  // 8 MB bf16
static constexpr size_t SZ_PH      = (size_t)BB * HEADS * NN * DH * 2; // 8 MB
static constexpr size_t OFF_SKIPF  = 0;
static constexpr size_t OFF_SKIPB  = OFF_SKIPF + SZ_SKIPF;
static constexpr size_t OFF_ARENA_A= OFF_SKIPB + SZ_SKIPB;          // qkvv bf16 -> conv_out f32
static constexpr size_t SZ_ARENA_A = (size_t)BB * NN * 1024 * 2;    // 32 MB
static constexpr size_t OFF_PH0    = OFF_ARENA_A + SZ_ARENA_A;      // qnT (later xsa_tmp)
static constexpr size_t OFF_PH1    = OFF_PH0 + SZ_PH;               // qnN (later h1)
static constexpr size_t OFF_PH2    = OFF_PH1 + SZ_PH;               // knN (later convf)
static constexpr size_t OFF_PH3    = OFF_PH2 + SZ_PH;               // kN
static constexpr size_t OFF_PH4    = OFF_PH3 + SZ_PH;               // vcaT
static constexpr size_t OFF_PH5    = OFF_PH4 + SZ_PH;               // vsaN
static constexpr size_t OFF_ARENA_B= OFF_PH5 + SZ_PH;               // xn bf16 -> attn_sa bf16 -> epa f32
static constexpr size_t SZ_ARENA_B = (size_t)BB * NN * CC * 4;      // 16 MB
static constexpr size_t OFF_XSA    = OFF_ARENA_B + SZ_ARENA_B;      // xsa_s bf16 (+ xca) -> conv8_out f32
static constexpr size_t OFF_XCA    = OFF_XSA + SZ_SKIPB;
static constexpr size_t OFF_CAL    = OFF_XCA + SZ_SKIPB;            // ca logits f32
static constexpr size_t OFF_ACA    = OFF_CAL + 131072;              // attn_ca bf16
static constexpr size_t OFF_KPJ    = OFF_ACA + 65536;               // k_projT bf16 (B,h,P,d)
static constexpr size_t OFF_VSP    = OFF_KPJ + 131072;              // v_sa_proj bf16 (B,h,d,P)
static constexpr size_t OFF_RNM    = OFF_VSP + 131072;              // rnorm f32 [2][1024]
static constexpr size_t OFF_BNS    = OFF_RNM + 8192;                // bn stats
static constexpr size_t OFF_WQ     = OFF_BNS + 2048;                // qkvv_w bf16
static constexpr size_t OFF_WEF    = OFF_WQ + 524288;               // ef_w bf16
static constexpr size_t OFF_WO1    = OFF_WEF + 524288;              // op1_w bf16
static constexpr size_t OFF_WO2    = OFF_WO1 + 65536;               // op2_w bf16
static constexpr size_t OFF_W1     = OFF_WO2 + 65536;               // conv1_w bf16
static constexpr size_t OFF_W2     = OFF_W1 + 3538944;              // conv2_w bf16
static constexpr size_t OFF_W8     = OFF_W2 + 3538944;              // conv8_w bf16

extern "C" void kernel_launch(void* const* d_in, const int* in_sizes, int n_in,
                              void* d_out, int out_size, void* d_ws, size_t ws_size,
                              hipStream_t stream) {
  (void)in_sizes; (void)n_in; (void)out_size; (void)ws_size;
  const float* x       = (const float*)d_in[0];
  const float* gamma   = (const float*)d_in[1];
  const float* ln_w    = (const float*)d_in[2];
  const float* ln_b    = (const float*)d_in[3];
  const float* qkvv_w  = (const float*)d_in[4];
  const float* ef_w    = (const float*)d_in[5];
  const float* ef_b    = (const float*)d_in[6];
  const float* temp    = (const float*)d_in[7];
  const float* temp2   = (const float*)d_in[8];
  const float* op1_w   = (const float*)d_in[9];
  const float* op1_b   = (const float*)d_in[10];
  const float* op2_w   = (const float*)d_in[11];
  const float* op2_b   = (const float*)d_in[12];
  const float* conv1_w = (const float*)d_in[13];
  const float* bn1_w   = (const float*)d_in[14];
  const float* bn1_b   = (const float*)d_in[15];
  const float* conv2_w = (const float*)d_in[16];
  const float* bn2_w   = (const float*)d_in[17];
  const float* bn2_b   = (const float*)d_in[18];
  const float* conv8_w = (const float*)d_in[19];
  const float* conv8_b = (const float*)d_in[20];
  float* out = (float*)d_out;

  char* w = (char*)d_ws;
  float*  skipF   = (float*)(w + OFF_SKIPF);
  __bf16* skipB   = (__bf16*)(w + OFF_SKIPB);
  __bf16* qkvv    = (__bf16*)(w + OFF_ARENA_A);
  float*  convOut = (float*)(w + OFF_ARENA_A);          // alias after qkvv dead
  __bf16* qnT     = (__bf16*)(w + OFF_PH0);
  __bf16* xsaTmp  = (__bf16*)(w + OFF_PH0);             // alias after qnT dead
  __bf16* qnN     = (__bf16*)(w + OFF_PH1);
  __bf16* h1      = (__bf16*)(w + OFF_PH1);             // alias (conv phase)
  __bf16* knN     = (__bf16*)(w + OFF_PH2);
  __bf16* convf   = (__bf16*)(w + OFF_PH2);             // alias (conv phase)
  __bf16* kN      = (__bf16*)(w + OFF_PH3);
  __bf16* vcaT    = (__bf16*)(w + OFF_PH4);
  __bf16* vsaN    = (__bf16*)(w + OFF_PH5);
  __bf16* xn      = (__bf16*)(w + OFF_ARENA_B);
  __bf16* attnSa  = (__bf16*)(w + OFF_ARENA_B);         // alias after xn dead
  float*  epa     = (float*)(w + OFF_ARENA_B);          // alias after attnSa dead
  __bf16* xsaS    = (__bf16*)(w + OFF_XSA);
  __bf16* xca     = (__bf16*)(w + OFF_XCA);
  float*  c8out   = (float*)(w + OFF_XSA);              // alias after op1/op2 done
  float*  caLog   = (float*)(w + OFF_CAL);
  __bf16* attnCa  = (__bf16*)(w + OFF_ACA);
  __bf16* kprojT  = (__bf16*)(w + OFF_KPJ);
  __bf16* vsaProj = (__bf16*)(w + OFF_VSP);
  float*  rnorm   = (float*)(w + OFF_RNM);
  float*  bnStats = (float*)(w + OFF_BNS);
  __bf16* wq  = (__bf16*)(w + OFF_WQ);
  __bf16* wef = (__bf16*)(w + OFF_WEF);
  __bf16* wo1 = (__bf16*)(w + OFF_WO1);
  __bf16* wo2 = (__bf16*)(w + OFF_WO2);
  __bf16* w1  = (__bf16*)(w + OFF_W1);
  __bf16* w2  = (__bf16*)(w + OFF_W2);
  __bf16* w8  = (__bf16*)(w + OFF_W8);

  auto cvt = [&](const float* src, __bf16* dst, long n) {
    cvt_f32_to_bf16<<<dim3((unsigned)((n + 255) / 256)), 256, 0, stream>>>(src, dst, n);
  };
  // Dispatcher: register-blocked 16x64 tiles when N allows, base 16x16 otherwise
  auto gemm = [&](const __bf16* A, const __bf16* Bm, const float* bias,
                  float* OF, __bf16* OB, int M, int Nd, int K,
                  int lda, int ldb, int ldo,
                  long sA, long sB, long sO1, long sO2, int oDiv,
                  int batch, int biasMode) {
    if (Nd % 64 == 0) {
      dim3 g(Nd / 64, M / 16, batch);
      wmma_gemm4_kernel<<<g, 32, 0, stream>>>(A, Bm, bias, OF, OB, K, lda, ldb,
                                              ldo, sA, sB, sO1, sO2, oDiv, biasMode);
    } else {
      dim3 g(Nd / 16, M / 16, batch);
      wmma_gemm_kernel<<<g, 32, 0, stream>>>(A, Bm, bias, OF, OB, K, lda, ldb,
                                             ldo, sA, sB, sO1, sO2, oDiv, biasMode);
    }
  };

  // --- weight conversions (cheap, deterministic each call) ---
  cvt(qkvv_w, wq, 1024L * 256);
  cvt(ef_w,   wef, 64L * 4096);
  cvt(op1_w,  wo1, 128L * 256);
  cvt(op2_w,  wo2, 128L * 256);
  cvt(conv1_w, w1, 256L * 256 * 27);
  cvt(conv2_w, w2, 256L * 256 * 27);
  cvt(conv8_w, w8, 256L * 256);

  const long BN = (long)BB * NN;      // 16384
  const long BH = (long)BB * HEADS;   // 32

  // --- LayerNorm -> xn (B,N,C) bf16 ---
  layernorm_kernel<<<dim3((unsigned)(BN / 256)), 256, 0, stream>>>(x, ln_w, ln_b, xn);

  // --- qkvv = xn @ W^T : (16384 x 1024), K=256 ---
  gemm(xn, wq, nullptr, nullptr, qkvv, (int)BN, 1024, 256, 256, 256, 1024,
       0, 0, 0, 0, 1, 1, 0);

  // --- q/k row norms + repack to per-head layouts ---
  qk_norm_kernel<<<2048, 256, 0, stream>>>(qkvv, rnorm);
  repack_kernel<<<dim3((unsigned)(BN * CC / 256)), 256, 0, stream>>>(
      qkvv, rnorm, qnT, qnN, knN, kN, vcaT, vsaN);

  // --- k_projT (B,h,P,d) = ef_w @ k^T + ef_b (per row) ---
  gemm(wef, kN, ef_b, nullptr, kprojT, PP, DH, NN, NN, NN, DH,
       0, (long)DH * NN, (long)PP * DH, 0, 1, (int)BH, 2);
  // --- v_sa_proj (B,h,d,P) = v_sa @ ef_w^T + ef_b (per col) ---
  gemm(vsaN, wef, ef_b, nullptr, vsaProj, DH, PP, NN, NN, NN, PP,
       (long)DH * NN, 0, (long)DH * PP, 0, 1, (int)BH, 1);

  // --- channel attention logits (B,h,32,32) = qn @ kn^T ---
  gemm(qnN, knN, nullptr, caLog, nullptr, DH, DH, NN, NN, NN, DH,
       (long)DH * NN, (long)DH * NN, (long)DH * DH, 0, 1, (int)BH, 0);
  softmax_ca_kernel<<<4, 256, 0, stream>>>(caLog, temp, attnCa);

  // --- x_ca (B,N,C) = (attn_ca @ v_ca)^T scattered to column block h*32 ---
  gemm(vcaT, attnCa, nullptr, nullptr, xca, NN, DH, DH, DH, DH, CC,
       (long)NN * DH, (long)DH * DH, (long)NN * CC, DH, HEADS, (int)BH, 0);

  // --- spatial attention logits (B,h,N,P) = qn^T @ k_projT^T -> bf16 ---
  gemm(qnT, kprojT, nullptr, nullptr, attnSa, NN, PP, DH, DH, DH, PP,
       (long)NN * DH, (long)PP * DH, (long)NN * PP, 0, 1, (int)BH, 0);
  softmax_sa_kernel<<<dim3((unsigned)(BH * NN / 256)), 256, 0, stream>>>(attnSa, temp2);

  // --- x_sa (B,h,N,d) = attn_sa @ v_sa_proj^T ---
  gemm(attnSa, vsaProj, nullptr, nullptr, xsaTmp, NN, DH, PP, PP, PP, DH,
       (long)NN * PP, (long)DH * PP, (long)NN * DH, 0, 1, (int)BH, 0);
  scramble_kernel<<<dim3((unsigned)(BN * CC / 256)), 256, 0, stream>>>(xsaTmp, xsaS);

  // --- op1 / op2 -> epa (B,N,C) f32 : cols [0,128) and [128,256) ---
  gemm(xsaS, wo1, op1_b, epa, nullptr, (int)BN, 128, 256, 256, 256, 256,
       0, 0, 0, 0, 1, 1, 1);
  gemm(xca, wo2, op2_b, epa + 128, nullptr, (int)BN, 128, 256, 256, 256, 256,
       0, 0, 0, 0, 1, 1, 1);

  // --- attn_skip = x + gamma * epa ---
  residual_kernel<<<dim3((unsigned)(BN * CC / 256)), 256, 0, stream>>>(
      x, gamma, epa, skipF, skipB);

  // --- UnetResBlock: conv1 -> bn1 -> lrelu -> conv2 -> bn2 -> +skip -> lrelu ---
  dim3 cgrid(2, (unsigned)(BN / 64));
  wmma_conv_kernel<<<cgrid, 256, 0, stream>>>(skipB, w1, convOut, CC, 3, 1, CC * 27);
  bn_stats_kernel<<<CC, 256, 0, stream>>>(convOut, bnStats);
  bn_lrelu_kernel<<<dim3((unsigned)(BN * CC / 256)), 256, 0, stream>>>(
      convOut, bnStats, bn1_w, bn1_b, h1);
  wmma_conv_kernel<<<cgrid, 256, 0, stream>>>(h1, w2, convOut, CC, 3, 1, CC * 27);
  bn_stats_kernel<<<CC, 256, 0, stream>>>(convOut, bnStats);
  bn_add_lrelu_kernel<<<dim3((unsigned)(BN * CC / 256)), 256, 0, stream>>>(
      convOut, bnStats, bn2_w, bn2_b, skipF, convf);

  // --- conv8 (1x1x1) + final residual ---
  wmma_conv_kernel<<<cgrid, 256, 0, stream>>>(convf, w8, c8out, CC, 1, 0, CC);
  final_kernel<<<dim3((unsigned)(BN * CC / 256)), 256, 0, stream>>>(
      skipF, c8out, conv8_b, out);
}
